// GCNBertSelfAttention_40836549050687
// MI455X (gfx1250) — matrix operand order repaired and checked by reading, not compiled, not run on hardware
//
#include <hip/hip_runtime.h>

// ---------------- problem constants ----------------
constexpr int kB = 8, kL = 256, kH = 768, kR = 128;
constexpr int kM = kB * kL;          // 2048 token rows
constexpr int kBase = 40;            // num_base_rels
constexpr int kRel = 80;

// ---------------- WMMA plumbing (gfx1250, wave32) ----------------
typedef __attribute__((ext_vector_type(16))) _Float16 v16h;
typedef __attribute__((ext_vector_type(8)))  float    v8f;
typedef unsigned int u32;

union Frag16 { u32 u[8]; v16h v; };

#define WMMA_F16(a, b, c) \
  __builtin_amdgcn_wmma_f32_16x16x32_f16(false, (a), false, (b), (short)0, (c), false, false)

// CDNA5 async copy: global -> LDS, 16B per lane, tracked by ASYNCcnt.
__device__ __forceinline__ void cp_async16(const _Float16* g, _Float16* l) {
  const unsigned loff = (unsigned)(unsigned long long)(void*)l;   // LDS offset = low 32 bits
  asm volatile("global_load_async_to_lds_b128 %0, %1, off" :: "v"(loff), "v"(g) : "memory");
}
template <int N>
__device__ __forceinline__ void wait_async() {
  asm volatile("s_wait_asynccnt %0" :: "n"(N) : "memory");
}

// A fragment (16x32 f16, M x K), per ISA 7.12.2:
//  lane = hf*16 + m ; VGPR p holds K = (p>=4?16:0) + hf*8 + 2*(p&3) .. +1
__device__ __forceinline__ v16h frag_a(const _Float16* lA, int ldh) {
  const int lane = threadIdx.x & 31;
  const int hf = lane >> 4, m = lane & 15;
  Frag16 f;
  const u32* row = (const u32*)(lA + m * ldh);
#pragma unroll
  for (int p = 0; p < 8; ++p) {
    const int k = ((p & 4) << 2) + hf * 8 + ((p & 3) << 1);
    f.u[p] = row[k >> 1];
  }
  return f.v;
}

// B fragment (32x16 f16, K x N): same K order as A: VGPR p holds K = (p>=4?16:0) + hf*8 + 2*(p&3) .. +1,
// N = lane&15.  Tile staged n-major in LDS: lB[n*ldh + k] = B(k, n).
__device__ __forceinline__ v16h frag_b(const _Float16* lB, int ldh) {
  const int lane = threadIdx.x & 31;
  const int hf = lane >> 4, n = lane & 15;
  Frag16 f;
  const u32* src = (const u32*)(lB + n * ldh);
#pragma unroll
  for (int p = 0; p < 8; ++p) f.u[p] = src[(((p & 4) << 2) + hf * 8 + ((p & 3) << 1)) >> 1];
  return f.v;
}

constexpr int BK = 32;   // K step per WMMA
constexpr int KP = 40;   // padded LDS stride (halfs); 80 B rows keep 16B alignment

// ---------------- kernel 1: f32 -> f16 convert ----------------
__global__ void cvt_f16(const float* __restrict__ s, _Float16* __restrict__ d, int n) {
  for (int i = blockIdx.x * blockDim.x + threadIdx.x; i < n; i += gridDim.x * blockDim.x)
    d[i] = (_Float16)s[i];
}

// ---------------- kernel 2: degrees + scaled f16 adjacency ----------------
__global__ __launch_bounds__(256) void prep_adj(const int* __restrict__ heads,
                                                _Float16* __restrict__ adjs,
                                                _Float16* __restrict__ adjTs,
                                                float* __restrict__ deg,
                                                float* __restrict__ rdeg) {
  const int row  = blockIdx.x * 8 + (threadIdx.x >> 5);
  const int lane = threadIdx.x & 31;
  const int b = row >> 8, i = row & 255;
  const int* hb = heads + b * kL * kL;
  int s = 0, sr = 0;
  for (int j = lane; j < kL; j += 32) { s += hb[i * kL + j]; sr += hb[j * kL + i]; }
#pragma unroll
  for (int off = 16; off; off >>= 1) { s += __shfl_xor(s, off, 32); sr += __shfl_xor(sr, off, 32); }
  const float d = (float)s, rd = (float)sr;
  const float inv  = 1.0f / ((d  > 0.f) ? d  : 1.f);
  const float rinv = 1.0f / ((rd > 0.f) ? rd : 1.f);
  if (lane == 0) { deg[row] = d; rdeg[row] = rd; }
  for (int j = lane; j < kL; j += 32) {
    adjs [row * kL + j] = (_Float16)(hb[i * kL + j] ? inv  : 0.f);
    adjTs[row * kL + j] = (_Float16)(hb[j * kL + i] ? rinv : 0.f);
  }
}

// ---------------- kernel 3: relation-embedding gather-sum ----------------
__global__ __launch_bounds__(256) void agg_kernel(const int* __restrict__ heads,
                                                  const int* __restrict__ rels,
                                                  const float* __restrict__ E,
                                                  _Float16* __restrict__ agg,
                                                  _Float16* __restrict__ ragg) {
  __shared__ float Elds[kRel * kR];                       // 40 KB
  for (int idx = threadIdx.x; idx < kRel * kR; idx += 256) Elds[idx] = E[idx];
  __syncthreads();

  const int row  = blockIdx.x * 8 + (threadIdx.x >> 5);
  const int lane = threadIdx.x & 31;
  const int b = row >> 8, i = row & 255;
  const int* hb = heads + b * kL * kL;
  const int* rb = rels  + b * kL * kL;
  const int base = lane * 4;
  float acc[4]  = {0.f, 0.f, 0.f, 0.f};
  float racc[4] = {0.f, 0.f, 0.f, 0.f};
  for (int j = 0; j < kL; ++j) {
    if (hb[i * kL + j]) {
      const float* e = Elds + rb[i * kL + j] * kR + base;
#pragma unroll
      for (int q = 0; q < 4; ++q) acc[q] += e[q];
    }
    if (hb[j * kL + i]) {
      const int rlt = rb[j * kL + i];
      if (rlt > 0) {
        const float* e = Elds + (rlt + kBase) * kR + base;
#pragma unroll
        for (int q = 0; q < 4; ++q) racc[q] += e[q];
      }
    }
  }
#pragma unroll
  for (int q = 0; q < 4; ++q) {
    agg [row * kR + base + q] = (_Float16)acc[q];
    ragg[row * kR + base + q] = (_Float16)racc[q];
  }
}

// ---------------- kernel 4: X @ W.T + bias (three projections) ----------------
// z=0: H1t = (hs@Wadj.T+badj)^T  [kH][kM] f16 ; z=1: H2t likewise
// z=2: out = hs@Wself.T+bself    [kM][kH] f32 (initializes d_out)
__global__ __launch_bounds__(256) void gemm_xw(const _Float16* __restrict__ X,
                                               const _Float16* __restrict__ W0,
                                               const _Float16* __restrict__ W1,
                                               const _Float16* __restrict__ W2,
                                               const float* __restrict__ b0,
                                               const float* __restrict__ b1,
                                               const float* __restrict__ b2,
                                               _Float16* __restrict__ H1t,
                                               _Float16* __restrict__ H2t,
                                               float* __restrict__ out) {
  __shared__ __align__(16) _Float16 lA[2 * 128 * KP];
  __shared__ __align__(16) _Float16 lB[2 * 64 * KP];
  const int n0 = blockIdx.x * 64, m0 = blockIdx.y * 128, z = blockIdx.z;
  const _Float16* W   = (z == 0) ? W0 : (z == 1) ? W1 : W2;
  const float*   bias = (z == 0) ? b0 : (z == 1) ? b1 : b2;
  const int tid = threadIdx.x, w = tid >> 5;
  v8f acc[4] = {};

  auto issue = [&](int k0, int buf) {                      // 3 async b128 per thread
    _Float16* dA = lA + buf * (128 * KP);
    _Float16* dB = lB + buf * (64 * KP);
#pragma unroll
    for (int t = 0; t < 2; ++t) {
      const int idx = tid + t * 256, r = idx >> 2, c = (idx & 3) * 8;
      cp_async16(X + (size_t)(m0 + r) * kH + k0 + c, dA + r * KP + c);
    }
    const int r = tid >> 2, c = (tid & 3) * 8;
    cp_async16(W + (size_t)(n0 + r) * kH + k0 + c, dB + r * KP + c);
  };

  constexpr int NS = kH / BK;                              // 24 K-steps
  issue(0, 0);
  for (int s = 0; s < NS; ++s) {
    const int cur = s & 1;
    __syncthreads();                                       // prev buffer fully consumed
    if (s + 1 < NS) { issue((s + 1) * BK, cur ^ 1); wait_async<3>(); }
    else            { wait_async<0>(); }
    __syncthreads();                                       // current buffer landed
    const _Float16* bA = lA + cur * (128 * KP);
    const _Float16* bB = lB + cur * (64 * KP);
    const v16h a = frag_a(bA + w * 16 * KP, KP);
#pragma unroll
    for (int nt = 0; nt < 4; ++nt) {
      const v16h b = frag_b(bB + nt * 16 * KP, KP);
      acc[nt] = WMMA_F16(a, b, acc[nt]);
    }
  }

  const int lane = tid & 31, hf = lane >> 4, col = lane & 15;
  const int mrow = m0 + w * 16 + hf * 8;
#pragma unroll
  for (int nt = 0; nt < 4; ++nt) {
    const int n = n0 + nt * 16 + col;
    const float bv = bias[n];
#pragma unroll
    for (int v = 0; v < 8; ++v) {
      const float val = acc[nt][v] + bv;
      const int m = mrow + v;
      if (z == 0)      H1t[(size_t)n * kM + m] = (_Float16)val;   // transposed
      else if (z == 1) H2t[(size_t)n * kM + m] = (_Float16)val;   // transposed
      else             out[(size_t)m * kH + n] = val;
    }
  }
}

// ---------------- kernel 5: out += adjs@H1 + adjTs@H2 (per batch) ----------------
__global__ __launch_bounds__(256) void gemm_ctx(const _Float16* __restrict__ adjs,
                                                const _Float16* __restrict__ adjTs,
                                                const _Float16* __restrict__ H1t,
                                                const _Float16* __restrict__ H2t,
                                                float* __restrict__ out) {
  __shared__ __align__(16) _Float16 lA1[2 * 128 * KP], lA2[2 * 128 * KP];
  __shared__ __align__(16) _Float16 lB1[2 * 64 * KP],  lB2[2 * 64 * KP];
  const int n0 = blockIdx.x * 64, m0 = blockIdx.y * 128, b = blockIdx.z;
  const int bL0 = b * kL;
  const _Float16* A1 = adjs  + (size_t)b * kL * kL;
  const _Float16* A2 = adjTs + (size_t)b * kL * kL;
  const int tid = threadIdx.x, w = tid >> 5;
  v8f acc[4] = {};

  auto issue = [&](int k0, int buf) {                      // 6 async b128 per thread
    _Float16* dA1 = lA1 + buf * (128 * KP);
    _Float16* dA2 = lA2 + buf * (128 * KP);
    _Float16* dB1 = lB1 + buf * (64 * KP);
    _Float16* dB2 = lB2 + buf * (64 * KP);
#pragma unroll
    for (int t = 0; t < 2; ++t) {
      const int idx = tid + t * 256, r = idx >> 2, c = (idx & 3) * 8;
      cp_async16(A1 + (size_t)(m0 + r) * kL + k0 + c, dA1 + r * KP + c);
      cp_async16(A2 + (size_t)(m0 + r) * kL + k0 + c, dA2 + r * KP + c);
    }
    const int r = tid >> 2, c = (tid & 3) * 8;
    cp_async16(H1t + (size_t)(n0 + r) * kM + bL0 + k0 + c, dB1 + r * KP + c);
    cp_async16(H2t + (size_t)(n0 + r) * kM + bL0 + k0 + c, dB2 + r * KP + c);
  };

  constexpr int NS = kL / BK;                              // 8 K-steps
  issue(0, 0);
  for (int s = 0; s < NS; ++s) {
    const int cur = s & 1;
    __syncthreads();
    if (s + 1 < NS) { issue((s + 1) * BK, cur ^ 1); wait_async<6>(); }
    else            { wait_async<0>(); }
    __syncthreads();
    const _Float16* bA1 = lA1 + cur * (128 * KP);
    const _Float16* bA2 = lA2 + cur * (128 * KP);
    const _Float16* bB1 = lB1 + cur * (64 * KP);
    const _Float16* bB2 = lB2 + cur * (64 * KP);
    const v16h a1 = frag_a(bA1 + w * 16 * KP, KP);
    const v16h a2 = frag_a(bA2 + w * 16 * KP, KP);
#pragma unroll
    for (int nt = 0; nt < 4; ++nt) {
      const v16h bb1 = frag_b(bB1 + nt * 16 * KP, KP);
      acc[nt] = WMMA_F16(a1, bb1, acc[nt]);
      const v16h bb2 = frag_b(bB2 + nt * 16 * KP, KP);
      acc[nt] = WMMA_F16(a2, bb2, acc[nt]);
    }
  }

  const int lane = tid & 31, hf = lane >> 4, col = lane & 15;
  const int mrow = bL0 + m0 + w * 16 + hf * 8;
#pragma unroll
  for (int nt = 0; nt < 4; ++nt) {
    const int n = n0 + nt * 16 + col;
#pragma unroll
    for (int v = 0; v < 8; ++v)
      out[(size_t)(mrow + v) * kH + n] += acc[nt][v];
  }
}

// ---------------- kernel 6: out += agg@Wr.T + ragg@Wrr.T + br*deg + brr*rdeg ------
__global__ __launch_bounds__(256) void gemm_rel(const _Float16* __restrict__ agg,
                                                const _Float16* __restrict__ ragg,
                                                const _Float16* __restrict__ Wr,
                                                const _Float16* __restrict__ Wrr,
                                                const float* __restrict__ br,
                                                const float* __restrict__ brr,
                                                const float* __restrict__ deg,
                                                const float* __restrict__ rdeg,
                                                float* __restrict__ out) {
  __shared__ __align__(16) _Float16 lA1[2 * 128 * KP], lA2[2 * 128 * KP];
  __shared__ __align__(16) _Float16 lB1[2 * 64 * KP],  lB2[2 * 64 * KP];
  const int n0 = blockIdx.x * 64, m0 = blockIdx.y * 128;
  const int tid = threadIdx.x, w = tid >> 5;
  v8f acc[4] = {};

  auto issue = [&](int k0, int buf) {                      // 6 async b128 per thread
    _Float16* dA1 = lA1 + buf * (128 * KP);
    _Float16* dA2 = lA2 + buf * (128 * KP);
    _Float16* dB1 = lB1 + buf * (64 * KP);
    _Float16* dB2 = lB2 + buf * (64 * KP);
#pragma unroll
    for (int t = 0; t < 2; ++t) {
      const int idx = tid + t * 256, r = idx >> 2, c = (idx & 3) * 8;
      cp_async16(agg  + (size_t)(m0 + r) * kR + k0 + c, dA1 + r * KP + c);
      cp_async16(ragg + (size_t)(m0 + r) * kR + k0 + c, dA2 + r * KP + c);
    }
    const int r = tid >> 2, c = (tid & 3) * 8;
    cp_async16(Wr  + (size_t)(n0 + r) * kR + k0 + c, dB1 + r * KP + c);
    cp_async16(Wrr + (size_t)(n0 + r) * kR + k0 + c, dB2 + r * KP + c);
  };

  constexpr int NS = kR / BK;                              // 4 K-steps
  issue(0, 0);
  for (int s = 0; s < NS; ++s) {
    const int cur = s & 1;
    __syncthreads();
    if (s + 1 < NS) { issue((s + 1) * BK, cur ^ 1); wait_async<6>(); }
    else            { wait_async<0>(); }
    __syncthreads();
    const _Float16* bA1 = lA1 + cur * (128 * KP);
    const _Float16* bA2 = lA2 + cur * (128 * KP);
    const _Float16* bB1 = lB1 + cur * (64 * KP);
    const _Float16* bB2 = lB2 + cur * (64 * KP);
    const v16h a1 = frag_a(bA1 + w * 16 * KP, KP);
    const v16h a2 = frag_a(bA2 + w * 16 * KP, KP);
#pragma unroll
    for (int nt = 0; nt < 4; ++nt) {
      const v16h bb1 = frag_b(bB1 + nt * 16 * KP, KP);
      acc[nt] = WMMA_F16(a1, bb1, acc[nt]);
      const v16h bb2 = frag_b(bB2 + nt * 16 * KP, KP);
      acc[nt] = WMMA_F16(a2, bb2, acc[nt]);
    }
  }

  const int lane = tid & 31, hf = lane >> 4, col = lane & 15;
  const int mrow = m0 + w * 16 + hf * 8;
#pragma unroll
  for (int nt = 0; nt < 4; ++nt) {
    const int n = n0 + nt * 16 + col;
    const float bn = br[n], bn2 = brr[n];
#pragma unroll
    for (int v = 0; v < 8; ++v) {
      const int m = mrow + v;
      out[(size_t)m * kH + n] += acc[nt][v] + bn * deg[m] + bn2 * rdeg[m];
    }
  }
}

// ---------------- host: workspace carving + launch sequence ----------------
extern "C" void kernel_launch(void* const* d_in, const int* in_sizes, int n_in,
                              void* d_out, int out_size, void* d_ws, size_t ws_size,
                              hipStream_t stream) {
  (void)in_sizes; (void)n_in; (void)out_size; (void)ws_size;
  const float* hs    = (const float*)d_in[0];
  const int*   heads = (const int*)  d_in[1];
  const int*   rels  = (const int*)  d_in[2];
  const float* E     = (const float*)d_in[3];
  const float* Wadj  = (const float*)d_in[4];
  const float* badj  = (const float*)d_in[5];
  const float* Wrev  = (const float*)d_in[6];
  const float* brev  = (const float*)d_in[7];
  const float* Wself = (const float*)d_in[8];
  const float* bself = (const float*)d_in[9];
  const float* Wr    = (const float*)d_in[10];
  const float* br    = (const float*)d_in[11];
  const float* Wrr   = (const float*)d_in[12];
  const float* brr   = (const float*)d_in[13];
  float* out = (float*)d_out;

  char* ws = (char*)d_ws;
  size_t off = 0;
  auto carve = [&](size_t bytes) -> void* {
    off = (off + 255) & ~(size_t)255;
    void* p = ws + off;
    off += bytes;
    return p;
  };
  _Float16* hs16    = (_Float16*)carve((size_t)kM * kH * 2);
  _Float16* Wadj16  = (_Float16*)carve((size_t)kH * kH * 2);
  _Float16* Wrev16  = (_Float16*)carve((size_t)kH * kH * 2);
  _Float16* Wself16 = (_Float16*)carve((size_t)kH * kH * 2);
  _Float16* Wr16    = (_Float16*)carve((size_t)kH * kR * 2);
  _Float16* Wrr16   = (_Float16*)carve((size_t)kH * kR * 2);
  _Float16* H1t     = (_Float16*)carve((size_t)kM * kH * 2);   // [kH][kM]
  _Float16* H2t     = (_Float16*)carve((size_t)kM * kH * 2);   // [kH][kM]
  _Float16* adjs    = (_Float16*)carve((size_t)kB * kL * kL * 2);
  _Float16* adjTs   = (_Float16*)carve((size_t)kB * kL * kL * 2);
  _Float16* agg16   = (_Float16*)carve((size_t)kM * kR * 2);
  _Float16* ragg16  = (_Float16*)carve((size_t)kM * kR * 2);
  float*    deg     = (float*)   carve((size_t)kM * 4);
  float*    rdeg    = (float*)   carve((size_t)kM * 4);

  // 1) precision converts (f32 -> f16 operands for the WMMA pipe)
  cvt_f16<<<512, 256, 0, stream>>>(hs,    hs16,    kM * kH);
  cvt_f16<<<512, 256, 0, stream>>>(Wadj,  Wadj16,  kH * kH);
  cvt_f16<<<512, 256, 0, stream>>>(Wrev,  Wrev16,  kH * kH);
  cvt_f16<<<512, 256, 0, stream>>>(Wself, Wself16, kH * kH);
  cvt_f16<<<256, 256, 0, stream>>>(Wr,    Wr16,    kH * kR);
  cvt_f16<<<256, 256, 0, stream>>>(Wrr,   Wrr16,   kH * kR);

  // 2) degrees + degree-normalized f16 adjacency (fwd + transposed)
  prep_adj<<<kM / 8, 256, 0, stream>>>(heads, adjs, adjTs, deg, rdeg);

  // 3) relation-embedding gather-sums (E cached in LDS)
  agg_kernel<<<kM / 8, 256, 0, stream>>>(heads, rels, E, agg16, ragg16);

  // 4) three H x H projections; z=2 initializes d_out with the self path
  gemm_xw<<<dim3(kH / 64, kM / 128, 3), 256, 0, stream>>>(
      hs16, Wadj16, Wrev16, Wself16, badj, brev, bself, H1t, H2t, out);

  // 5) out += adjacency-mean context (fwd + rev fused, 1/deg folded into A)
  gemm_ctx<<<dim3(kH / 64, kL / 128, kB), 256, 0, stream>>>(adjs, adjTs, H1t, H2t, out);

  // 6) out += relation path + degree-scaled biases
  gemm_rel<<<dim3(kH / 64, kM / 128), 256, 0, stream>>>(
      agg16, ragg16, Wr16, Wrr16, br, brr, deg, rdeg, out);
}
